// VectorQuantizer_67353677136168
// MI455X (gfx1250) — matrix operand order repaired and checked by reading, not compiled, not run on hardware
//
#include <hip/hip_runtime.h>
#include <hip/hip_bf16.h>

#define N_ROWS 32768
#define DIM    256
#define KCB    8192

typedef __attribute__((ext_vector_type(16))) __bf16 v16bf;
typedef __attribute__((ext_vector_type(8)))  __bf16 v8bf;
typedef __attribute__((ext_vector_type(8)))  float  v8f;

__device__ __forceinline__ __bf16 f2bf(float x) { return (__bf16)x; }

#define WMMA_BF16(A, B, C) \
  __builtin_amdgcn_wmma_f32_16x16x32_bf16(false, (A), false, (B), (short)0, (C), false, false)

// ---------------------------------------------------------------------------
// Kernel 1: split codebook into hi/lo bf16 and compute ||w||^2 per row.
// One wave32 per codebook row; 8 rows per 256-thread block.
// ---------------------------------------------------------------------------
__global__ void vq_prep_w(const float* __restrict__ W,
                          __bf16* __restrict__ Whi, __bf16* __restrict__ Wlo,
                          float* __restrict__ wnorm) {
  const int wave = threadIdx.x >> 5;
  const int lane = threadIdx.x & 31;
  const int row  = blockIdx.x * 8 + wave;
  const float* wr = W + (size_t)row * DIM;
  float ss = 0.f;
#pragma unroll
  for (int t = 0; t < DIM / 32; ++t) {
    const int d = lane + t * 32;
    const float x = wr[d];
    ss += x * x;
    const __bf16 h = f2bf(x);
    const __bf16 l = f2bf(x - (float)h);
    Whi[(size_t)row * DIM + d] = h;
    Wlo[(size_t)row * DIM + d] = l;
  }
#pragma unroll
  for (int m = 16; m >= 1; m >>= 1) ss += __shfl_xor(ss, m, 32);
  if (lane == 0) wnorm[row] = ss;
}

// ---------------------------------------------------------------------------
// Kernel 2: streaming argmin via bf16x3 WMMA GEMM.
// Block = 128 threads (4 waves); block owns 64 rows of X, staged hi/lo in LDS
// (A-frags become register-resident across the whole K loop). Each wave
// streams TWO 16-codeword chunks per iteration into two independent
// accumulator chains so dependent-WMMA latency is hidden.
// dist(n,k) = ||w_k||^2 - 2 * x_n . w_k   (||x||^2 dropped: row-constant).
// ---------------------------------------------------------------------------
__global__ void __launch_bounds__(128)
vq_argmin(const float* __restrict__ X,
          const __bf16* __restrict__ Whi, const __bf16* __restrict__ Wlo,
          const float* __restrict__ wnorm,
          int* __restrict__ idxOut) {
  __shared__ __bf16 sXhi[64 * DIM];   // 32 KB
  __shared__ __bf16 sXlo[64 * DIM];   // 32 KB

  const int tid     = threadIdx.x;
  const int lane    = tid & 31;
  const int wave    = tid >> 5;
  const int rowBase = blockIdx.x * 64;

  // Stage + hi/lo split the 64x256 X tile.
  for (int i = tid; i < 64 * DIM; i += 128) {
    const float x = X[(size_t)rowBase * DIM + i];
    const __bf16 h = f2bf(x);
    sXhi[i] = h;
    sXlo[i] = f2bf(x - (float)h);
  }
  __syncthreads();

  const int col     = lane & 15;        // B/C column this lane holds
  const int halfsel = lane >> 4;        // which 16-lane half
  const int arow    = wave * 16 + col;  // A row this lane sources (ISA A layout)
  const __bf16* xh = &sXhi[(size_t)arow * DIM];
  const __bf16* xl = &sXlo[(size_t)arow * DIM];

  float minv0[8], minv1[8];
  int   mini0[8], mini1[8];
#pragma unroll
  for (int j = 0; j < 8; ++j) {
    minv0[j] = 3.4e38f; mini0[j] = 0;
    minv1[j] = 3.4e38f; mini1[j] = 0;
  }

  for (int kc = 0; kc < KCB; kc += 32) {
    const int r0 = kc + col;        // chunk 0 codebook row for this lane
    const int r1 = kc + 16 + col;   // chunk 1 codebook row
    const __bf16* wh0 = Whi + (size_t)r0 * DIM + halfsel * 16;
    const __bf16* wl0 = Wlo + (size_t)r0 * DIM + halfsel * 16;
    const __bf16* wh1 = Whi + (size_t)r1 * DIM + halfsel * 16;
    const __bf16* wl1 = Wlo + (size_t)r1 * DIM + halfsel * 16;

    // Near-scope prefetch of the next 2-chunk pair (16 KB per array).
    if (kc + 32 < KCB) {
      const size_t pf = (size_t)(kc + 32) * DIM;
      __builtin_prefetch(Whi + pf + lane * 128, 0, 3);
      __builtin_prefetch(Whi + pf + 4096 + lane * 128, 0, 3);
      __builtin_prefetch(Wlo + pf + lane * 128, 0, 3);
      __builtin_prefetch(Wlo + pf + 4096 + lane * 128, 0, 3);
    }

    const float wn0 = wnorm[r0];
    const float wn1 = wnorm[r1];
    v8f acc0 = {};
    v8f acc1 = {};
#pragma unroll
    for (int d0 = 0; d0 < DIM; d0 += 32) {
      // A frag (16x32 bf16): lane holds K = d0+half*16+[0..15].
      // (Loop-invariant over kc -> compiler keeps these in registers.)
      v8bf a0h = *(const v8bf*)(xh + d0 + halfsel * 16);
      v8bf a1h = *(const v8bf*)(xh + d0 + 8 + halfsel * 16);
      v8bf a0l = *(const v8bf*)(xl + d0 + halfsel * 16);
      v8bf a1l = *(const v8bf*)(xl + d0 + 8 + halfsel * 16);
      v16bf ah = __builtin_shufflevector(a0h, a1h, 0,1,2,3,4,5,6,7,8,9,10,11,12,13,14,15);
      v16bf al = __builtin_shufflevector(a0l, a1l, 0,1,2,3,4,5,6,7,8,9,10,11,12,13,14,15);
      // B frags (32x16 bf16): lane = column, 16 consecutive K per lane-half.
      v16bf bh0 = *(const v16bf*)(wh0 + d0);
      v16bf bl0 = *(const v16bf*)(wl0 + d0);
      v16bf bh1 = *(const v16bf*)(wh1 + d0);
      v16bf bl1 = *(const v16bf*)(wl1 + d0);

      // Two independent accumulation chains, interleaved.
      acc0 = WMMA_BF16(ah, bh0, acc0);
      acc1 = WMMA_BF16(ah, bh1, acc1);
      acc0 = WMMA_BF16(ah, bl0, acc0);
      acc1 = WMMA_BF16(ah, bl1, acc1);
      acc0 = WMMA_BF16(al, bh0, acc0);
      acc1 = WMMA_BF16(al, bh1, acc1);
    }

#pragma unroll
    for (int j = 0; j < 8; ++j) {
      const float d0v = wn0 - 2.0f * acc0[j];
      const float d1v = wn1 - 2.0f * acc1[j];
      if (d0v < minv0[j]) { minv0[j] = d0v; mini0[j] = r0; }
      if (d1v < minv1[j]) { minv1[j] = d1v; mini1[j] = r1; }
    }
  }

  // Merge the two chunk-streams, then min+index reduce across the 16 lanes
  // that share each C row.
#pragma unroll
  for (int j = 0; j < 8; ++j) {
    float v  = minv0[j];
    int   ix = mini0[j];
    if (minv1[j] < v || (minv1[j] == v && mini1[j] < ix)) { v = minv1[j]; ix = mini1[j]; }
#pragma unroll
    for (int m = 1; m <= 8; m <<= 1) {
      const float ov = __shfl_xor(v, m, 32);
      const int   oi = __shfl_xor(ix, m, 32);
      if (ov < v || (ov == v && oi < ix)) { v = ov; ix = oi; }
    }
    if (col == 0) {
      const int row = rowBase + wave * 16 + j + 8 * halfsel;
      idxOut[row] = ix;
    }
  }
}

// ---------------------------------------------------------------------------
// Kernel 3: gather quantized rows out[n,:] = W[idx[n],:] with float4 copies.
// ---------------------------------------------------------------------------
__global__ void vq_gather(const float* __restrict__ W,
                          const int* __restrict__ idx,
                          float* __restrict__ out) {
  const int gid = blockIdx.x * blockDim.x + threadIdx.x;  // one per 4 floats
  const int n   = gid >> 6;      // DIM/4 = 64 float4 per row
  const int d4  = gid & 63;
  const int k   = idx[n];
  const float4* src = (const float4*)(W + (size_t)k * DIM) + d4;
  float4* dst       = (float4*)(out + (size_t)n * DIM) + d4;
  *dst = *src;
}

// ---------------------------------------------------------------------------
extern "C" void kernel_launch(void* const* d_in, const int* in_sizes, int n_in,
                              void* d_out, int out_size, void* d_ws, size_t ws_size,
                              hipStream_t stream) {
  const float* X = (const float*)d_in[0];   // [N, D] fp32
  const float* W = (const float*)d_in[1];   // [K, D] fp32

  float* outQ   = (float*)d_out;                         // [N, D] fp32
  int*   outIdx = (int*)(outQ + (size_t)N_ROWS * DIM);   // [N] int32 (tuple tail)

  // Workspace layout (~8.03 MB): Whi | Wlo | wnorm
  __bf16* Whi   = (__bf16*)d_ws;
  __bf16* Wlo   = Whi + (size_t)KCB * DIM;
  float*  wnorm = (float*)(Wlo + (size_t)KCB * DIM);

  vq_prep_w<<<KCB / 8, 256, 0, stream>>>(W, Whi, Wlo, wnorm);
  vq_argmin<<<N_ROWS / 64, 128, 0, stream>>>(X, Whi, Wlo, wnorm, outIdx);
  vq_gather<<<(N_ROWS * (DIM / 4)) / 256, 256, 0, stream>>>(W, outIdx, outQ);
}
